// LovaszLoss_52321291600338
// MI455X (gfx1250) — matrix operand order repaired and checked by reading, not compiled, not run on hardware
//
#include <hip/hip_runtime.h>
#include <hip/hip_bf16.h>

typedef __attribute__((ext_vector_type(16))) _Float16 v16h;
typedef __attribute__((ext_vector_type(8)))  float    v8f;

#define NCLS 20
#define RADIX 256
#define SORT_BLOCK 4096
#define SORT_THREADS 256
#define SORT_ROUNDS 16
#define SEG 4096
#define SEG_ROUNDS 16

static __device__ __forceinline__ unsigned ballot32(int p) {
    return (unsigned)__ballot(p);
}

// ---------------- zero accumulators (d_ws is poisoned, not re-zeroed) ----------
__global__ void zero_kernel(int* gts, float* perClass) {
    int t = threadIdx.x;
    if (t < NCLS) gts[t] = 0;
    if (t >= 32 && t < 32 + NCLS) perClass[t - 32] = 0.0f;
}

// ---------------- fused softmax + key generation ------------------------------
// key[c*N + n] = (bits(|1 - softmax(preds[n])[c]|) << 1) | (labels[n] == c)
__global__ __launch_bounds__(256) void keygen_kernel(
        const float* __restrict__ preds, const int* __restrict__ labels,
        unsigned* __restrict__ keys, int N) {
    int n = blockIdx.x * blockDim.x + threadIdx.x;
    if (n >= N) return;
    float x[NCLS];
    const float4* row = reinterpret_cast<const float4*>(preds + (size_t)n * NCLS);
    #pragma unroll
    for (int i = 0; i < NCLS / 4; ++i) {
        float4 v = row[i];
        x[4*i+0] = v.x; x[4*i+1] = v.y; x[4*i+2] = v.z; x[4*i+3] = v.w;
    }
    float m = x[0];
    #pragma unroll
    for (int c = 1; c < NCLS; ++c) m = fmaxf(m, x[c]);
    float s = 0.0f;
    #pragma unroll
    for (int c = 0; c < NCLS; ++c) s += __expf(x[c] - m);
    float inv = 1.0f / s;
    int lab = labels[n];
    #pragma unroll
    for (int c = 0; c < NCLS; ++c) {
        float p = __expf(x[c] - m) * inv;
        float e = fabsf(1.0f - p);                     // e in [0,1] -> bits < 0x3F800001
        unsigned key = (__float_as_uint(e) << 1) | (unsigned)(lab == c);
        keys[(size_t)c * N + n] = key;
    }
}

// ---------------- label histogram (gts per class) -----------------------------
__global__ void gts_kernel(const int* __restrict__ labels, int* gts, int N) {
    __shared__ int h[NCLS];
    if (threadIdx.x < NCLS) h[threadIdx.x] = 0;
    __syncthreads();
    for (int i = blockIdx.x * blockDim.x + threadIdx.x; i < N;
         i += gridDim.x * blockDim.x) {
        int l = labels[i];
        if (l >= 0 && l < NCLS) atomicAdd(&h[l], 1);
    }
    __syncthreads();
    if (threadIdx.x < NCLS) atomicAdd(&gts[threadIdx.x], h[threadIdx.x]);
}

// ---------------- radix pass 1/3: per-block digit histograms ------------------
__global__ __launch_bounds__(SORT_THREADS) void hist_kernel(
        const unsigned* __restrict__ src, unsigned* __restrict__ histG,
        int N, int NBLK, int shift) {
    int blk = blockIdx.x, c = blockIdx.y, t = threadIdx.x;
    __shared__ unsigned h[RADIX];
    h[t] = 0;
    __syncthreads();
    for (int r = 0; r < SORT_ROUNDS; ++r) {
        int i = blk * SORT_BLOCK + r * SORT_THREADS + t;
        if (i < N) {
            unsigned key = src[(size_t)c * N + i];
            atomicAdd(&h[(key >> shift) & 255u], 1u);
        }
    }
    __syncthreads();
    histG[((size_t)c * NBLK + blk) * RADIX + t] = h[t];
}

// ---------------- radix pass 2/3: scan -> block-excl prefixes + DESC bases ----
__global__ __launch_bounds__(RADIX) void scan_kernel(
        unsigned* __restrict__ histG, unsigned* __restrict__ digitBase, int NBLK) {
    int c = blockIdx.x, d = threadIdx.x;
    unsigned run = 0;
    for (int b = 0; b < NBLK; ++b) {
        size_t idx = ((size_t)c * NBLK + b) * RADIX + d;
        unsigned v = histG[idx];
        histG[idx] = run;                 // block-exclusive prefix per digit
        run += v;
    }
    __shared__ unsigned tot[RADIX];
    __shared__ unsigned baseS[RADIX];
    tot[d] = run;
    __syncthreads();
    if (d == 0) {                         // descending digit order -> desc sort
        unsigned acc = 0;
        for (int dd = RADIX - 1; dd >= 0; --dd) { baseS[dd] = acc; acc += tot[dd]; }
    }
    __syncthreads();
    digitBase[c * RADIX + d] = baseS[d];
}

// ---------------- radix pass 3/3: stable scatter (wave32 ballot ranking) ------
__global__ __launch_bounds__(SORT_THREADS) void scatter_kernel(
        const unsigned* __restrict__ src, unsigned* __restrict__ dst,
        const unsigned* __restrict__ histG, const unsigned* __restrict__ digitBase,
        int N, int NBLK, int shift) {
    int blk = blockIdx.x, c = blockIdx.y, t = threadIdx.x;
    int lane = t & 31, wave = t >> 5;
    unsigned ltmask = (1u << lane) - 1u;
    __shared__ unsigned run[RADIX];
    __shared__ unsigned wcnt[8 * RADIX];
    run[t] = digitBase[c * RADIX + t] + histG[((size_t)c * NBLK + blk) * RADIX + t];
    for (int r = 0; r < SORT_ROUNDS; ++r) {
        __syncthreads();                               // run init / prev round safe
        #pragma unroll
        for (int j = 0; j < 8; ++j) wcnt[j * RADIX + t] = 0;
        __syncthreads();
        int i = blk * SORT_BLOCK + r * SORT_THREADS + t;
        bool valid = i < N;
        unsigned key = valid ? src[(size_t)c * N + i] : 0u;
        unsigned digit = (key >> shift) & 255u;
        unsigned vm = ballot32(valid);
        unsigned peers = vm;
        #pragma unroll
        for (int bit = 0; bit < 8; ++bit) {
            unsigned b = ballot32((int)((digit >> bit) & 1u));
            peers &= ((digit >> bit) & 1u) ? b : ~b;
        }
        unsigned rankw = __popc(peers & ltmask);
        if (valid && rankw == 0) wcnt[wave * RADIX + digit] = __popc(peers);
        __syncthreads();
        if (valid) {
            unsigned ofs = run[digit];
            for (int w = 0; w < wave; ++w) ofs += wcnt[w * RADIX + digit];
            ofs += rankw;
            dst[(size_t)c * N + ofs] = key;
        }
        __syncthreads();
        unsigned add = 0;
        #pragma unroll
        for (int j = 0; j < 8; ++j) add += wcnt[j * RADIX + t];
        run[t] += add;
    }
}

// ---------------- reduction phase 1: per-segment fg totals --------------------
__global__ __launch_bounds__(256) void seg_sum_kernel(
        const unsigned* __restrict__ keys, int* __restrict__ segTot, int N, int NSEG) {
    int seg = blockIdx.x, c = blockIdx.y, t = threadIdx.x;
    const unsigned* k = keys + (size_t)c * N;
    int sum = 0;
    for (int r = 0; r < SEG_ROUNDS; ++r) {
        int i = seg * SEG + r * 256 + t;
        if (i < N) sum += (int)(k[i] & 1u);
    }
    __shared__ int red[256];
    red[t] = sum;
    __syncthreads();
    for (int s = 128; s > 0; s >>= 1) {
        if (t < s) red[t] += red[t + s];
        __syncthreads();
    }
    if (t == 0) segTot[c * NSEG + seg] = red[0];
}

// ---------------- reduction phase 2: exclusive scan of segment totals ---------
__global__ void seg_scan_kernel(const int* __restrict__ segTot,
                                int* __restrict__ segCarry, int NSEG) {
    int c = blockIdx.x;
    if (threadIdx.x == 0) {
        int run = 0;
        for (int s = 0; s < NSEG; ++s) {
            segCarry[c * NSEG + s] = run;
            run += segTot[c * NSEG + s];
        }
    }
}

// ---------------- reduction phase 3: Lovasz terms with WMMA prefix-scan -------
// per-class loss = sum_i J_i * (e_i - e_{i+1}),
// J_i = 1 - (G - cum_i) / (G + (i+1) - cum_i), cum_i = inclusive fg prefix
__global__ __launch_bounds__(256) void seg_reduce_kernel(
        const unsigned* __restrict__ keys, const int* __restrict__ gts,
        const int* __restrict__ segCarry, float* __restrict__ perClass,
        int N, int NSEG) {
    int seg = blockIdx.x, c = blockIdx.y, t = threadIdx.x, lane = t & 31;
    __shared__ _Float16 fgS[256];
    __shared__ float scanS[256];
    __shared__ float eS[256];
    __shared__ float red[256];
    const unsigned* k = keys + (size_t)c * N;
    float G = (float)gts[c];
    float carry = (float)segCarry[c * NSEG + seg];
    float sum = 0.0f;
    for (int chunk = 0; chunk < SEG_ROUNDS; ++chunk) {
        int i = seg * SEG + chunk * 256 + t;
        bool valid = i < N;
        unsigned key = valid ? k[i] : 0u;
        float fg = valid ? (float)(key & 1u) : 0.0f;
        float e  = valid ? __uint_as_float(key >> 1) : 0.0f;
        fgS[t] = (_Float16)fg;
        eS[t]  = e;
        __syncthreads();
        if (t < 32) {                                   // wave0, EXEC all ones
            int m = lane & 15;
            v16h a, b;
            #pragma unroll
            for (int j = 0; j < 16; ++j) { a[j] = (_Float16)0.0f; b[j] = (_Float16)0.0f; }
            int kofs = (lane < 16) ? 0 : 16;            // A 16x32 f16, contiguous k = 16h + s
            #pragma unroll
            for (int j = 0; j < 16; ++j)
                a[j] = (_Float16)((j + kofs <= m) ? 1.0f : 0.0f);  // lower-tri ones
            if (lane < 16) {                            // B rows K<16 = fg matrix
                #pragma unroll
                for (int j = 0; j < 16; ++j)
                    b[j] = fgS[m * 16 + j];             // F[kk][n] = fg[n*16+kk], n==lane
            }
            v8f d = {};
            d = __builtin_amdgcn_wmma_f32_16x16x32_f16(
                    false, a, false, b, (short)0, d, false, false);
            int mofs = (lane < 16) ? 0 : 8;
            int n = lane & 15;
            #pragma unroll
            for (int r = 0; r < 8; ++r)
                scanS[n * 16 + (r + mofs)] = d[r];      // inclusive prefix per column
        }
        __syncthreads();
        int ncol = t >> 4;
        float colofs = 0.0f, totalAll = 0.0f;
        for (int n2 = 0; n2 < 16; ++n2) {
            float ct = scanS[n2 * 16 + 15];
            if (n2 < ncol) colofs += ct;
            totalAll += ct;
        }
        float cum = carry + colofs + scanS[t];          // inclusive fg prefix count
        int nexti = seg * SEG + chunk * 256 + 256;
        float e_next = (t < 255) ? eS[t + 1]
                     : ((nexti < N) ? __uint_as_float(k[nexti] >> 1) : 0.0f);
        if (valid) {
            float rank1 = (float)(i + 1);
            float inter = G - cum;
            float uni   = G + rank1 - cum;              // >= 1 always
            float J = 1.0f - inter / uni;
            sum += J * (e - e_next);
        }
        carry += totalAll;
        __syncthreads();
    }
    red[t] = sum;
    __syncthreads();
    for (int s = 128; s > 0; s >>= 1) {
        if (t < s) red[t] += red[t + s];
        __syncthreads();
    }
    if (t == 0) atomicAdd(&perClass[c], red[0]);
}

// ---------------- classes='present' mean --------------------------------------
__global__ void finalize_kernel(const float* __restrict__ perClass,
                                const int* __restrict__ gts, float* out) {
    if (threadIdx.x == 0) {
        float s = 0.0f, n = 0.0f;
        for (int c = 0; c < NCLS; ++c) {
            if (gts[c] > 0) { s += perClass[c]; n += 1.0f; }
        }
        out[0] = s / fmaxf(n, 1.0f);
    }
}

extern "C" void kernel_launch(void* const* d_in, const int* in_sizes, int n_in,
                              void* d_out, int out_size, void* d_ws, size_t ws_size,
                              hipStream_t stream) {
    const float* preds  = (const float*)d_in[0];
    const int*   labels = (const int*)d_in[1];
    float* out = (float*)d_out;
    int N = in_sizes[1];
    int NBLK = (N + SORT_BLOCK - 1) / SORT_BLOCK;
    int NSEG = (N + SEG - 1) / SEG;

    unsigned char* ws = (unsigned char*)d_ws;
    size_t keyBytes  = (size_t)NCLS * N * sizeof(unsigned);
    size_t histBytes = (size_t)NCLS * NBLK * RADIX * sizeof(unsigned);
    size_t segBytes  = (size_t)NCLS * NSEG * sizeof(int);
    size_t off = 0;
    unsigned* keysA     = (unsigned*)(ws + off); off += keyBytes;
    unsigned* keysB     = (unsigned*)(ws + off); off += keyBytes;
    unsigned* histG     = (unsigned*)(ws + off); off += histBytes;
    unsigned* digitBase = (unsigned*)(ws + off); off += (size_t)NCLS * RADIX * 4;
    int*      gts       = (int*)     (ws + off); off += (size_t)NCLS * 4;
    int*      segTot    = (int*)     (ws + off); off += segBytes;
    int*      segCarry  = (int*)     (ws + off); off += segBytes;
    float*    perClass  = (float*)   (ws + off); off += (size_t)NCLS * 4;

    zero_kernel<<<1, 64, 0, stream>>>(gts, perClass);
    keygen_kernel<<<(N + 255) / 256, 256, 0, stream>>>(preds, labels, keysA, N);
    gts_kernel<<<256, 256, 0, stream>>>(labels, gts, N);

    unsigned* src = keysA;
    unsigned* dst = keysB;
    for (int pass = 0; pass < 4; ++pass) {
        int shift = pass * 8;
        hist_kernel<<<dim3(NBLK, NCLS), SORT_THREADS, 0, stream>>>(src, histG, N, NBLK, shift);
        scan_kernel<<<NCLS, RADIX, 0, stream>>>(histG, digitBase, NBLK);
        scatter_kernel<<<dim3(NBLK, NCLS), SORT_THREADS, 0, stream>>>(
            src, dst, histG, digitBase, N, NBLK, shift);
        unsigned* tmp = src; src = dst; dst = tmp;
    }
    // after 4 passes sorted keys are back in keysA (== src)
    seg_sum_kernel<<<dim3(NSEG, NCLS), 256, 0, stream>>>(src, segTot, N, NSEG);
    seg_scan_kernel<<<NCLS, 32, 0, stream>>>(segTot, segCarry, NSEG);
    seg_reduce_kernel<<<dim3(NSEG, NCLS), 256, 0, stream>>>(
        src, gts, segCarry, perClass, N, NSEG);
    finalize_kernel<<<1, 32, 0, stream>>>(perClass, gts, out);
}
